// GINTopK2_72095321030886
// MI455X (gfx1250) — matrix operand order repaired and checked by reading, not compiled, not run on hardware
//
#include <hip/hip_runtime.h>
#include <math.h>

typedef __attribute__((ext_vector_type(8)))  float v8f;
typedef __attribute__((ext_vector_type(16))) __bf16 v16bf;
typedef __attribute__((ext_vector_type(4)))  unsigned int v4u;
typedef __attribute__((ext_vector_type(8)))  unsigned int v8u;

#define HID 256

__device__ __forceinline__ float gelu_f(float x) {
    const float c0 = 0.7978845608028654f;  // sqrt(2/pi)
    float x3 = x * x * x;
    return 0.5f * x * (1.0f + tanhf(c0 * (x + 0.044715f * x3)));
}

__device__ __forceinline__ unsigned int bf16bits(float f) {
    unsigned int u = __float_as_uint(f);
    return (u + 0x7FFFu + ((u >> 16) & 1u)) >> 16;   // round-to-nearest-even
}

__device__ __forceinline__ int imin(int a, int b) { return a < b ? a : b; }

// ---------------------------------------------------------------------------
// Pack f32 weights (kdim x 256) into bf16 B-fragment dwords:
// pw[(k/2)*256 + n] = bf16(w[k+1][n])<<16 | bf16(w[k][n])
// ---------------------------------------------------------------------------
__global__ __launch_bounds__(256) void pack_w_kernel(
    const float* __restrict__ w, unsigned int* __restrict__ pw, int kdim)
{
    int idx = blockIdx.x * 256 + threadIdx.x;
    if (idx >= (kdim / 2) * HID) return;
    int krow = idx >> 8, n = idx & 255;
    float lo = w[(size_t)(2 * krow) * HID + n];
    float hi = w[(size_t)(2 * krow + 1) * HID + n];
    pw[idx] = (bf16bits(hi) << 16) | bf16bits(lo);
}

// ---------------------------------------------------------------------------
// Per-graph LDS scatter-aggregation (edges are grouped by graph):
//   lagg[dst_local][col] += x[src][col]  via ds_add_f32, then stream out.
// nid != nullptr => layer-2: remap through newid, skip dropped edges.
// ---------------------------------------------------------------------------
template<int NODES, int FDIM, int COLS>
__global__ __launch_bounds__(256) void agg_lds_kernel(
    const float* __restrict__ x, const int* __restrict__ src,
    const int* __restrict__ dst, const int* __restrict__ nid,
    float* __restrict__ agg, int epg)
{
    extern __shared__ float lagg[];            // NODES * COLS floats
    const int nchunks = FDIM / COLS;
    const int g = blockIdx.x / nchunks;
    const int c0 = (blockIdx.x % nchunks) * COLS;

    for (int i = threadIdx.x; i < NODES * COLS; i += 256) lagg[i] = 0.f;
    __syncthreads();

    const int ebase = g * epg;
    for (int i = threadIdx.x; i < epg * COLS; i += 256) {
        int e = ebase + (i / COLS);
        int col = i & (COLS - 1);
        int s = src[e], d = dst[e];
        if (nid) {
            s = nid[s]; d = nid[d];
            if ((s | d) < 0) continue;         // dropped edge
        }
        float v = x[(size_t)s * FDIM + c0 + col];
        atomicAdd(&lagg[(d - g * NODES) * COLS + col], v);
    }
    __syncthreads();

    for (int i = threadIdx.x; i < NODES * COLS; i += 256) {
        int n = i / COLS, col = i & (COLS - 1);
        agg[(size_t)(g * NODES + n) * FDIM + c0 + col] = lagg[i];
    }
}

// ---------------------------------------------------------------------------
// Fused GIN MLP: y = gelu( gelu((x+agg) @ wa + ba) @ wb + bb )
// 64 rows/block, 8 waves, bf16 WMMA 16x16x32 with f32 accumulate.
// Weights pre-packed (pack_w_kernel); A/Mid staged as bf16 in LDS.
// ---------------------------------------------------------------------------
template<int FIN>
__global__ __launch_bounds__(256) void mlp_kernel(
    const float* __restrict__ xin, const float* __restrict__ agg,
    const unsigned int* __restrict__ pwa, const float* __restrict__ ba,
    const unsigned int* __restrict__ pwb, const float* __restrict__ bb,
    float* __restrict__ yout)
{
    __shared__ unsigned short As[64 * FIN];
    __shared__ unsigned short Mid[64 * HID];

    const int tid  = threadIdx.x;
    const int lane = tid & 31;
    const int wave = tid >> 5;
    const int row0 = blockIdx.x * 64;

    // Stage A = bf16(x + agg)
    for (int i = tid; i < 64 * FIN; i += 256) {
        int r = i / FIN, c = i & (FIN - 1);
        float v = xin[(size_t)(row0 + r) * FIN + c] + agg[(size_t)(row0 + r) * FIN + c];
        As[i] = (unsigned short)bf16bits(v);
    }
    __syncthreads();

    const int m16    = lane & 15;
    const int kh     = lane >> 4;     // lane half: element s holds k = 16*kh + s (contiguous k order)
    const int rowsel = kh * 8;        // C/D fragment M offset

    // Phase A: Mid = gelu(As @ wa + ba)   (64 x FIN) @ (FIN x 256)
    for (int t = wave; t < 64; t += 8) {
        int tm = t >> 4, tn = t & 15;
        int col = tn * 16 + m16;
        const unsigned short* arow = &As[(tm * 16 + m16) * FIN];
        v8f acc = {};
        for (int k0 = 0; k0 < FIN; k0 += 32) {
            v4u lo = *(const v4u*)(arow + k0 + kh * 16);
            v4u hi = *(const v4u*)(arow + k0 + kh * 16 + 8);
            v8u au = __builtin_shufflevector(lo, hi, 0, 1, 2, 3, 4, 5, 6, 7);
            v16bf a = __builtin_bit_cast(v16bf, au);
            v8u bu;
            int bbase = ((k0 >> 1) + kh * 8) * HID + col;
#pragma unroll
            for (int gg = 0; gg < 8; ++gg) bu[gg] = pwa[bbase + gg * HID];
            v16bf b = __builtin_bit_cast(v16bf, bu);
            acc = __builtin_amdgcn_wmma_f32_16x16x32_bf16(
                false, a, false, b, (short)0, acc, false, false);
        }
        float bias = ba[col];
#pragma unroll
        for (int r = 0; r < 8; ++r) {
            float v = gelu_f(acc[r] + bias);
            Mid[(tm * 16 + rowsel + r) * HID + col] = (unsigned short)bf16bits(v);
        }
    }
    __syncthreads();

    // Phase B: y = gelu(Mid @ wb + bb)    (64 x 256) @ (256 x 256)
    for (int t = wave; t < 64; t += 8) {
        int tm = t >> 4, tn = t & 15;
        int col = tn * 16 + m16;
        const unsigned short* arow = &Mid[(tm * 16 + m16) * HID];
        v8f acc = {};
        for (int k0 = 0; k0 < HID; k0 += 32) {
            v4u lo = *(const v4u*)(arow + k0 + kh * 16);
            v4u hi = *(const v4u*)(arow + k0 + kh * 16 + 8);
            v8u au = __builtin_shufflevector(lo, hi, 0, 1, 2, 3, 4, 5, 6, 7);
            v16bf a = __builtin_bit_cast(v16bf, au);
            v8u bu;
            int bbase = ((k0 >> 1) + kh * 8) * HID + col;
#pragma unroll
            for (int gg = 0; gg < 8; ++gg) bu[gg] = pwb[bbase + gg * HID];
            v16bf b = __builtin_bit_cast(v16bf, bu);
            acc = __builtin_amdgcn_wmma_f32_16x16x32_bf16(
                false, a, false, b, (short)0, acc, false, false);
        }
        float bias = bb[col];
#pragma unroll
        for (int r = 0; r < 8; ++r) {
            float v = gelu_f(acc[r] + bias);
            yout[(size_t)(row0 + tm * 16 + rowsel + r) * HID + col] = v;
        }
    }
}

// ---------------------------------------------------------------------------
// BatchNorm: column sums (atomics), finalize into y*A + C, apply
// ---------------------------------------------------------------------------
__global__ __launch_bounds__(256) void bn_stats_kernel(
    const float* __restrict__ y, float* __restrict__ sum,
    float* __restrict__ sumsq, int nrows)
{
    int c  = threadIdx.x;
    int r0 = blockIdx.x * 256;
    int r1 = imin(r0 + 256, nrows);
    float s = 0.f, q = 0.f;
    for (int r = r0; r < r1; ++r) {
        float v = y[(size_t)r * HID + c];
        s += v; q += v * v;
    }
    atomicAdd(&sum[c], s);
    atomicAdd(&sumsq[c], q);
}

__global__ __launch_bounds__(256) void bn_finalize_kernel(
    const float* __restrict__ sum, const float* __restrict__ sumsq,
    const float* __restrict__ g, const float* __restrict__ be,
    float* __restrict__ A, float* __restrict__ C, float invN)
{
    int c = threadIdx.x;
    float m = sum[c] * invN;
    float v = sumsq[c] * invN - m * m;
    float a = rsqrtf(v + 1e-5f) * g[c];
    A[c] = a;
    C[c] = be[c] - m * a;
}

__global__ __launch_bounds__(256) void bn_apply_kernel(
    float* __restrict__ y, const float* __restrict__ A,
    const float* __restrict__ C, size_t n)
{
    size_t i = (size_t)blockIdx.x * 256 + threadIdx.x;
    if (i >= n) return;
    int c = (int)(i & (HID - 1));
    y[i] = y[i] * A[c] + C[c];
}

// ---------------------------------------------------------------------------
// Per-node score: tanh(dot(y_i, pw) / ||pw||)  (one block per node)
// ---------------------------------------------------------------------------
__global__ __launch_bounds__(256) void score_kernel(
    const float* __restrict__ y, const float* __restrict__ pw,
    float* __restrict__ score)
{
    __shared__ float red[256];
    __shared__ float red2[256];
    int i = blockIdx.x, c = threadIdx.x;
    float w = pw[c];
    red[c]  = y[(size_t)i * HID + c] * w;
    red2[c] = w * w;
    __syncthreads();
    for (int s = 128; s > 0; s >>= 1) {
        if (c < s) { red[c] += red[c + s]; red2[c] += red2[c + s]; }
        __syncthreads();
    }
    if (c == 0) score[i] = tanhf(red[0] * rsqrtf(red2[0]));
}

// ---------------------------------------------------------------------------
// Per-graph top-k via bitonic sort in LDS (descending); writes perm/val/newid
// ---------------------------------------------------------------------------
template<int NODES, int THREADS>
__global__ __launch_bounds__(THREADS) void topk_kernel(
    const float* __restrict__ scores, int* __restrict__ newid,
    int* __restrict__ perm, float* __restrict__ val, int k)
{
    __shared__ float s[NODES];
    __shared__ int   id[NODES];
    int g = blockIdx.x;
    for (int i = threadIdx.x; i < NODES; i += THREADS) {
        s[i]  = scores[(size_t)g * NODES + i];
        id[i] = i;
    }
    __syncthreads();
    for (int size = 2; size <= NODES; size <<= 1) {
        for (int stride = size >> 1; stride > 0; stride >>= 1) {
            for (int i = threadIdx.x; i < NODES; i += THREADS) {
                int j = i ^ stride;
                if (j > i) {
                    bool desc = ((i & size) == 0);
                    float si = s[i], sj = s[j];
                    bool sw = desc ? (si < sj) : (si > sj);
                    if (sw) {
                        s[i] = sj; s[j] = si;
                        int t = id[i]; id[i] = id[j]; id[j] = t;
                    }
                }
            }
            __syncthreads();
        }
    }
    for (int r = threadIdx.x; r < k; r += THREADS) {
        int oldn = g * NODES + id[r];
        int nn   = g * k + r;
        newid[oldn] = nn;
        perm[nn]    = oldn;
        val[nn]     = s[r];
    }
}

__global__ __launch_bounds__(256) void gather_kernel(
    const float* __restrict__ y, const int* __restrict__ perm,
    const float* __restrict__ val, float* __restrict__ xp)
{
    int j = blockIdx.x, c = threadIdx.x;
    xp[(size_t)j * HID + c] = y[(size_t)perm[j] * HID + c] * val[j];
}

// readout: per graph [max over k rows, mean over k rows] -> 512 cols
__global__ __launch_bounds__(256) void readout_kernel(
    const float* __restrict__ xp, float* __restrict__ out, int k)
{
    int g = blockIdx.x, c = threadIdx.x;
    const float* base = xp + (size_t)g * k * HID;
    float mx = -INFINITY, sm = 0.f;
    for (int r = 0; r < k; ++r) {
        float v = base[(size_t)r * HID + c];
        mx = fmaxf(mx, v);
        sm += v;
    }
    out[(size_t)g * 512 + c]       = mx;
    out[(size_t)g * 512 + HID + c] = sm / (float)k;
}

// out = (x1 + x2) @ lw + lb      (128 x 512) @ (512 x 256)
__global__ __launch_bounds__(256) void final_kernel(
    const float* __restrict__ x1, const float* __restrict__ x2,
    const float* __restrict__ lw, const float* __restrict__ lb,
    float* __restrict__ out)
{
    int b = blockIdx.x, n = threadIdx.x;
    float acc = lb[n];
    const float* r1 = x1 + (size_t)b * 512;
    const float* r2 = x2 + (size_t)b * 512;
    for (int k = 0; k < 512; ++k)
        acc += (r1[k] + r2[k]) * lw[(size_t)k * HID + n];
    out[(size_t)b * HID + n] = acc;
}

// ---------------------------------------------------------------------------
extern "C" void kernel_launch(void* const* d_in, const int* in_sizes, int n_in,
                              void* d_out, int out_size, void* d_ws, size_t ws_size,
                              hipStream_t stream) {
    const float* x   = (const float*)d_in[0];
    const int*   ei  = (const int*)d_in[1];
    const float* w1a = (const float*)d_in[3];
    const float* b1a = (const float*)d_in[4];
    const float* w1b = (const float*)d_in[5];
    const float* b1b = (const float*)d_in[6];
    const float* g1  = (const float*)d_in[7];
    const float* be1 = (const float*)d_in[8];
    const float* pw1 = (const float*)d_in[9];
    const float* w2a = (const float*)d_in[10];
    const float* b2a = (const float*)d_in[11];
    const float* w2b = (const float*)d_in[12];
    const float* b2b = (const float*)d_in[13];
    const float* g2  = (const float*)d_in[14];
    const float* be2 = (const float*)d_in[15];
    const float* pw2 = (const float*)d_in[16];
    const float* lw  = (const float*)d_in[17];
    const float* lb  = (const float*)d_in[18];
    float* out = (float*)d_out;

    const int F  = 128, NPG = 1024;
    const int N1 = in_sizes[0] / F;          // 131072
    const int B  = N1 / NPG;                 // 128
    const int E  = in_sizes[1] / 2;          // 524288
    const int EPG = E / B;                   // 4096
    const int K1 = NPG / 2;                  // 512
    const int N2 = B * K1;                   // 65536
    const int K2 = K1 / 2;                   // 256
    const int N3 = B * K2;                   // 32768
    const int* srcI = ei;
    const int* dstI = ei + E;

    char* ws = (char*)d_ws;
    float* y    = (float*)(ws);                       // N1*256 f32 (128MiB), reused as y2
    float* agg  = (float*)(ws + (size_t)134217728);   // 64MiB (N1*128 or N2*256)
    float* xp1  = (float*)(ws + (size_t)201326592);   // N2*256 (64MiB)
    float* xp2  = (float*)(ws + (size_t)268435456);   // N3*256 (32MiB)
    float* scb  = (float*)(ws + (size_t)301989888);   // scores (N1 f32)
    int*   prm  = (int*)  (ws + (size_t)302514176);   // perm (N2 int)
    float* vlb  = (float*)(ws + (size_t)302776320);   // val  (N2 f32)
    int*   nid  = (int*)  (ws + (size_t)303038464);   // newid (N1 int)
    unsigned int* pw1a = (unsigned int*)(ws + (size_t)303562752);  // 64KiB
    unsigned int* pw1b = (unsigned int*)(ws + (size_t)303628288);  // 128KiB
    unsigned int* pw2a = (unsigned int*)(ws + (size_t)303759360);  // 128KiB
    unsigned int* pw2b = (unsigned int*)(ws + (size_t)303890432);  // 128KiB
    float* ssum = (float*)(ws + (size_t)309854208);
    float* ssq  = ssum + 256;
    float* bnA  = ssum + 512;
    float* bnC  = ssum + 768;
    float* x1   = (float*)(ws + (size_t)309858304);   // B*512
    float* x2   = (float*)(ws + (size_t)310120448);   // B*512

    // ---- Pack weights to bf16 fragment layout ----
    pack_w_kernel<<<64, 256, 0, stream>>>(w1a, pw1a, 128);
    pack_w_kernel<<<128, 256, 0, stream>>>(w1b, pw1b, 256);
    pack_w_kernel<<<128, 256, 0, stream>>>(w2a, pw2a, 256);
    pack_w_kernel<<<128, 256, 0, stream>>>(w2b, pw2b, 256);

    // ---- Layer 1: GIN conv ----
    agg_lds_kernel<1024, 128, 32><<<B * 4, 256, 1024 * 32 * sizeof(float), stream>>>(
        x, srcI, dstI, nullptr, agg, EPG);
    mlp_kernel<128><<<N1 / 64, 256, 0, stream>>>(x, agg, pw1a, b1a, pw1b, b1b, y);

    // ---- BN 1 ----
    hipMemsetAsync(ssum, 0, 2 * HID * sizeof(float), stream);
    bn_stats_kernel<<<(N1 + 255) / 256, 256, 0, stream>>>(y, ssum, ssq, N1);
    bn_finalize_kernel<<<1, 256, 0, stream>>>(ssum, ssq, g1, be1, bnA, bnC, 1.0f / N1);
    bn_apply_kernel<<<(int)(((size_t)N1 * HID + 255) / 256), 256, 0, stream>>>(
        y, bnA, bnC, (size_t)N1 * HID);

    // ---- TopK pool 1 ----
    score_kernel<<<N1, 256, 0, stream>>>(y, pw1, scb);
    hipMemsetAsync(nid, 0xFF, (size_t)N1 * sizeof(int), stream);
    topk_kernel<1024, 512><<<B, 512, 0, stream>>>(scb, nid, prm, vlb, K1);
    gather_kernel<<<N2, 256, 0, stream>>>(y, prm, vlb, xp1);
    readout_kernel<<<B, 256, 0, stream>>>(xp1, x1, K1);

    // ---- Layer 2: GIN conv on pooled graph (remap via nid on the fly) ----
    agg_lds_kernel<512, 256, 64><<<B * 4, 256, 512 * 64 * sizeof(float), stream>>>(
        xp1, srcI, dstI, nid, agg, EPG);
    mlp_kernel<256><<<N2 / 64, 256, 0, stream>>>(xp1, agg, pw2a, b2a, pw2b, b2b, y);

    // ---- BN 2 ----
    hipMemsetAsync(ssum, 0, 2 * HID * sizeof(float), stream);
    bn_stats_kernel<<<(N2 + 255) / 256, 256, 0, stream>>>(y, ssum, ssq, N2);
    bn_finalize_kernel<<<1, 256, 0, stream>>>(ssum, ssq, g2, be2, bnA, bnC, 1.0f / N2);
    bn_apply_kernel<<<(int)(((size_t)N2 * HID + 255) / 256), 256, 0, stream>>>(
        y, bnA, bnC, (size_t)N2 * HID);

    // ---- TopK pool 2 (no conv afterwards; edges not needed) ----
    score_kernel<<<N2, 256, 0, stream>>>(y, pw2, scb);
    hipMemsetAsync(nid, 0xFF, (size_t)N2 * sizeof(int), stream);
    topk_kernel<512, 256><<<B, 256, 0, stream>>>(scb, nid, prm, vlb, K2);
    gather_kernel<<<N3, 256, 0, stream>>>(y, prm, vlb, xp2);
    readout_kernel<<<B, 256, 0, stream>>>(xp2, x2, K2);

    // ---- Final linear ----
    final_kernel<<<B, 256, 0, stream>>>(x1, x2, lw, lb, out);
}
